// SphericalStructureLearner_61624190763410
// MI455X (gfx1250) — matrix operand order repaired and checked by reading, not compiled, not run on hardware
//
#include <hip/hip_runtime.h>
#include <hip/hip_bf16.h>

typedef _Float16 v16h __attribute__((ext_vector_type(16)));
typedef _Float16 v8h  __attribute__((ext_vector_type(8)));
typedef float    v8f  __attribute__((ext_vector_type(8)));

#define WMMA_F16(a, b, c) \
  __builtin_amdgcn_wmma_f32_16x16x32_f16(false, (a), false, (b), (short)0, (c), false, false)

// ---- A-fragment loader: lane holds row (lane&15), K = {d0..d0+15} ----
__device__ __forceinline__ v16h load_a_h(const _Float16* p) {
  v8h lo = *(const v8h*)(p);
  v8h hi = *(const v8h*)(p + 8);
  v16h r;
#pragma unroll
  for (int i = 0; i < 8; ++i) { r[i] = lo[i]; r[i + 8] = hi[i]; }
  return r;
}
__device__ __forceinline__ v16h make_a(v8h lo, v8h hi) {
  v16h r;
#pragma unroll
  for (int i = 0; i < 8; ++i) { r[i] = lo[i]; r[i + 8] = hi[i]; }
  return r;
}

// =============== W: f32 -> f16 ===============
__global__ void k_wcvt(const float* __restrict__ src, _Float16* __restrict__ dst, int n) {
  int i = blockIdx.x * 256 + threadIdx.x;
  if (i < n) dst[i] = (_Float16)src[i];
}

// =============== Y = normalize_rows(X @ W^T), f16 [M,256] out ===============
// Transposed-D form: A = W-rows (n x d), B = X-rows as columns (d x m).
// D tile is (n x m): lane = output row m, the 8 acc regs = 8 consecutive n -> b128 stores.
template <typename T>
__global__ void __launch_bounds__(256) k_linproj(const T* __restrict__ X,
                                                 const _Float16* __restrict__ W,
                                                 _Float16* __restrict__ Y, int M) {
  const int lane = threadIdx.x & 31;
  const int wv   = threadIdx.x >> 5;
  const int lr   = lane & 15;
  const int hi   = lane >> 4;  // 0/1
  const int row0 = blockIdx.x * 128 + wv * 16;

  int ra = row0 + lr; if (ra > M - 1) ra = M - 1;  // clamped row for loads

  v8f acc[16];
#pragma unroll
  for (int nt = 0; nt < 16; ++nt) acc[nt] = (v8f)(0.0f);

#pragma unroll
  for (int ks = 0; ks < 8; ++ks) {
    // B-fragment: 16 contiguous K(=d) of row ra
    const int dB = ks * 32 + hi * 16;
    v16h b;
    if constexpr (sizeof(T) == 4) {
      v8f x0 = *(const v8f*)((const float*)X + (size_t)ra * 256 + dB);
      v8f x1 = *(const v8f*)((const float*)X + (size_t)ra * 256 + dB + 8);
#pragma unroll
      for (int i = 0; i < 8; ++i) { b[i] = (_Float16)x0[i]; b[i + 8] = (_Float16)x1[i]; }
    } else {
      b = *(const v16h*)((const _Float16*)X + (size_t)ra * 256 + dB);
    }
    const int dA = ks * 32 + hi * 16;
#pragma unroll
    for (int nt = 0; nt < 16; ++nt) {
      const int n = nt * 16 + lr;
      v16h a = load_a_h(W + (size_t)n * 256 + dA);
      acc[nt] = WMMA_F16(a, b, acc[nt]);
    }
  }
  // row norm: lane owns row (row0+lr); halves hold complementary column sets
  float s = 0.f;
#pragma unroll
  for (int nt = 0; nt < 16; ++nt)
#pragma unroll
    for (int r = 0; r < 8; ++r) s += acc[nt][r] * acc[nt][r];
  s += __shfl_xor(s, 16);
  const float sc = rsqrtf(s + 1e-30f);
  const int row = row0 + lr;
  if (row < M) {
#pragma unroll
    for (int nt = 0; nt < 16; ++nt) {
      v8h p;
#pragma unroll
      for (int r = 0; r < 8; ++r) p[r] = (_Float16)(acc[nt][r] * sc);
      *(v8h*)(Y + (size_t)row * 256 + nt * 16 + 8 * hi) = p;
    }
  }
}

// =============== Phase-1a: E[g,s,t] = exp(q_s . k_t - 1), f16, row-major [s][t] ===============
// Transposed-D: A = gathered k-rows (t x d), B = gathered q-row as column (d x s).
// Lane = s; acc regs = consecutive t -> packed b128 stores of E rows.
__global__ void __launch_bounds__(256) k_scores(const _Float16* __restrict__ Yq,
                                                const _Float16* __restrict__ Yk,
                                                const int* __restrict__ subsets,
                                                _Float16* __restrict__ Eall) {
  const int g = blockIdx.x;
  const int lane = threadIdx.x & 31, wv = threadIdx.x >> 5;
  const int lr = lane & 15, hi = lane >> 4;

  const int qa = subsets[g * 128 + wv * 16 + lr];  // q row (B column)
  int kb[8];
#pragma unroll
  for (int tt = 0; tt < 8; ++tt) kb[tt] = subsets[g * 128 + tt * 16 + lr];  // k rows (A)

  v8f acc[8];
#pragma unroll
  for (int tt = 0; tt < 8; ++tt) acc[tt] = (v8f)(0.0f);

#pragma unroll
  for (int ks = 0; ks < 8; ++ks) {
    v16h b = *(const v16h*)(Yq + (size_t)qa * 256 + ks * 32 + hi * 16);
    const int dA = ks * 32 + hi * 16;
#pragma unroll
    for (int tt = 0; tt < 8; ++tt) {
      v16h a = load_a_h(Yk + (size_t)kb[tt] * 256 + dA);
      acc[tt] = WMMA_F16(a, b, acc[tt]);
    }
  }
  const size_t sg = (size_t)g * 128 + wv * 16 + lr;  // this lane's s row
#pragma unroll
  for (int tt = 0; tt < 8; ++tt) {
    v8h p;
#pragma unroll
    for (int r = 0; r < 8; ++r) p[r] = (_Float16)__expf(acc[tt][r] - 1.0f);
    *(v8h*)(Eall + sg * 128 + tt * 16 + 8 * hi) = p;
  }
}

// =============== Phase-1b: att[s,:] = normalize(E[s,:] @ V), f16 ===============
// Transposed-D: A = V^T rows from LDS (d x t, XOR-swizzled), B = E rows (t x s).
// Lane = s; acc regs = consecutive d -> packed b128 stores.
__global__ void __launch_bounds__(256) k_attn_av(const _Float16* __restrict__ Eall,
                                                 const _Float16* __restrict__ Yv,
                                                 const int* __restrict__ subsets,
                                                 _Float16* __restrict__ att) {
  __shared__ _Float16 Vt[256 * 128];  // [d][t ^ ((d&7)*16)], 64KB
  const int g = blockIdx.x;
  const int tid = threadIdx.x;
  const int lane = tid & 31, wv = tid >> 5;
  const int lr = lane & 15, hi = lane >> 4;

  // gather V rows, transpose into swizzled LDS (lanes sweep t -> ~2-way banked)
  {
    const int t = tid & 127;
    const int node = subsets[g * 128 + t];
#pragma unroll
    for (int it = 0; it < 16; ++it) {
      const int ch = it * 2 + (tid >> 7);
      v8h v = *(const v8h*)(Yv + (size_t)node * 256 + ch * 8);
#pragma unroll
      for (int i = 0; i < 8; ++i) {
        const int d = ch * 8 + i;
        Vt[d * 128 + (t ^ ((d & 7) * 16))] = v[i];
      }
    }
  }
  __syncthreads();

  v8f acc[16];
#pragma unroll
  for (int dt = 0; dt < 16; ++dt) acc[dt] = (v8f)(0.0f);

  const size_t sg = (size_t)g * 128 + wv * 16 + lr;  // this lane's s row
#pragma unroll
  for (int ks = 0; ks < 4; ++ks) {
    v16h b = *(const v16h*)(Eall + sg * 128 + ks * 32 + hi * 16);
    const int t0 = ks * 32 + hi * 16;
#pragma unroll
    for (int dt = 0; dt < 16; ++dt) {
      const int d = dt * 16 + lr;
      const int sw = (d & 7) * 16;
      v8h a0 = *(const v8h*)(&Vt[d * 128 + (t0 ^ sw)]);
      v8h a1 = *(const v8h*)(&Vt[d * 128 + ((t0 + 8) ^ sw)]);
      acc[dt] = WMMA_F16(make_a(a0, a1), b, acc[dt]);
    }
  }
  float s = 0.f;
#pragma unroll
  for (int dt = 0; dt < 16; ++dt)
#pragma unroll
    for (int r = 0; r < 8; ++r) s += acc[dt][r] * acc[dt][r];
  s += __shfl_xor(s, 16);
  const float sc = rsqrtf(s + 1e-30f);
#pragma unroll
  for (int dt = 0; dt < 16; ++dt) {
    v8h p;
#pragma unroll
    for (int r = 0; r < 8; ++r) p[r] = (_Float16)(acc[dt][r] * sc);
    *(v8h*)(att + sg * 256 + dt * 16 + 8 * hi) = p;
  }
}

// =============== Phase-2 self term: out[n,:] = exp(q2.k_self - 1) * v_self ===============
__global__ void k_self(const _Float16* __restrict__ q2, const _Float16* __restrict__ ks,
                       const _Float16* __restrict__ vs, float* __restrict__ out, int N) {
  const int gid = blockIdx.x * 8 + (threadIdx.x >> 5);
  if (gid >= N) return;
  const int lane = threadIdx.x & 31;
  v8h q = *(const v8h*)(q2 + (size_t)gid * 256 + lane * 8);
  v8h k = *(const v8h*)(ks + (size_t)gid * 256 + lane * 8);
  float s = 0.f;
#pragma unroll
  for (int i = 0; i < 8; ++i) s += (float)q[i] * (float)k[i];
  s += __shfl_xor(s, 1); s += __shfl_xor(s, 2); s += __shfl_xor(s, 4);
  s += __shfl_xor(s, 8); s += __shfl_xor(s, 16);
  const float e = __expf(s - 1.0f);
  v8h v = *(const v8h*)(vs + (size_t)gid * 256 + lane * 8);
#pragma unroll
  for (int i = 0; i < 8; ++i) out[(size_t)gid * 256 + lane * 8 + i] = e * (float)v[i];
}

// =============== Phase-2 occurrence scatter ===============
__global__ void k_occ(const _Float16* __restrict__ q2, const _Float16* __restrict__ ko,
                      const _Float16* __restrict__ vo, const int* __restrict__ seg,
                      float* __restrict__ out, int M) {
  const int j = blockIdx.x * 8 + (threadIdx.x >> 5);
  if (j >= M) return;
  const int lane = threadIdx.x & 31;
  const int node = seg[j];
  v8h q = *(const v8h*)(q2 + (size_t)node * 256 + lane * 8);
  v8h k = *(const v8h*)(ko + (size_t)j * 256 + lane * 8);
  float s = 0.f;
#pragma unroll
  for (int i = 0; i < 8; ++i) s += (float)q[i] * (float)k[i];
  s += __shfl_xor(s, 1); s += __shfl_xor(s, 2); s += __shfl_xor(s, 4);
  s += __shfl_xor(s, 8); s += __shfl_xor(s, 16);
  const float e = __expf(s - 1.0f);
  v8h v = *(const v8h*)(vo + (size_t)j * 256 + lane * 8);
#pragma unroll
  for (int i = 0; i < 8; ++i)
    unsafeAtomicAdd(&out[(size_t)node * 256 + lane * 8 + i], e * (float)v[i]);
}

// =============== final row normalize (softmax denom cancels under proj) ===============
__global__ void k_norm(float* __restrict__ out, int N) {
  const int gid = blockIdx.x * 8 + (threadIdx.x >> 5);
  if (gid >= N) return;
  const int lane = threadIdx.x & 31;
  float x[8];
  float s = 0.f;
#pragma unroll
  for (int i = 0; i < 8; ++i) { x[i] = out[(size_t)gid * 256 + lane * 8 + i]; s += x[i] * x[i]; }
  s += __shfl_xor(s, 1); s += __shfl_xor(s, 2); s += __shfl_xor(s, 4);
  s += __shfl_xor(s, 8); s += __shfl_xor(s, 16);
  const float sc = rsqrtf(s + 1e-30f);
#pragma unroll
  for (int i = 0; i < 8; ++i) out[(size_t)gid * 256 + lane * 8 + i] = x[i] * sc;
}

extern "C" void kernel_launch(void* const* d_in, const int* in_sizes, int n_in,
                              void* d_out, int out_size, void* d_ws, size_t ws_size,
                              hipStream_t stream) {
  const float* x_S     = (const float*)d_in[0];
  const int*   subsets = (const int*)d_in[1];
  const float* Wsrc[6] = {(const float*)d_in[2], (const float*)d_in[3], (const float*)d_in[4],
                          (const float*)d_in[5], (const float*)d_in[6], (const float*)d_in[7]};
  const int N  = in_sizes[0] / 256;
  const int MS = in_sizes[1];       // G*S occurrences
  const int G  = MS / 128;
  float* out = (float*)d_out;

  // ---- workspace carve-up (f16 buffers) ----
  char* ws = (char*)d_ws;
  size_t off = 0;
  auto take = [&](size_t bytes) { char* p = ws + off; off = (off + bytes + 255) & ~(size_t)255; return p; };
  _Float16* Wf[6];
  for (int i = 0; i < 6; ++i) Wf[i] = (_Float16*)take((size_t)65536 * 2);
  _Float16* att  = (_Float16*)take((size_t)MS * 256 * 2);
  _Float16* Eall = (_Float16*)take((size_t)G * 128 * 128 * 2);
  _Float16* Y1q  = (_Float16*)take((size_t)N * 256 * 2);
  _Float16* Y1k  = (_Float16*)take((size_t)N * 256 * 2);
  _Float16* Y1v  = (_Float16*)take((size_t)N * 256 * 2);
  _Float16* kocc = (_Float16*)take((size_t)MS * 256 * 2);
  _Float16* vocc = (_Float16*)take((size_t)MS * 256 * 2);
  // Y1q/Y1k/Y1v dead after phase 1 -> alias for q2/k_self/v_self
  _Float16* q2 = Y1q; _Float16* kslf = Y1k; _Float16* vslf = Y1v;

  // 1) weights to f16
  for (int i = 0; i < 6; ++i)
    k_wcvt<<<(65536 + 255) / 256, 256, 0, stream>>>(Wsrc[i], Wf[i], 65536);

  const int rowBlocksN = (N + 127) / 128;
  const int rowBlocksM = (MS + 127) / 128;

  // 2) phase-1 linears for every node (proj commutes with gather)
  k_linproj<float><<<rowBlocksN, 256, 0, stream>>>(x_S, Wf[0], Y1q, N);
  k_linproj<float><<<rowBlocksN, 256, 0, stream>>>(x_S, Wf[1], Y1k, N);
  k_linproj<float><<<rowBlocksN, 256, 0, stream>>>(x_S, Wf[2], Y1v, N);

  // 3) per-group exp(scores-1)  (constant shift valid: |q.k| <= 1)
  k_scores<<<G, 256, 0, stream>>>(Y1q, Y1k, subsets, Eall);

  // 4) per-group E @ V, normalized (softmax denom cancels under proj)
  k_attn_av<<<G, 256, 0, stream>>>(Eall, Y1v, subsets, att);

  // 5) phase-2 linears
  k_linproj<float><<<rowBlocksN, 256, 0, stream>>>(x_S, Wf[3], q2,   N);
  k_linproj<float><<<rowBlocksN, 256, 0, stream>>>(x_S, Wf[4], kslf, N);
  k_linproj<float><<<rowBlocksN, 256, 0, stream>>>(x_S, Wf[5], vslf, N);
  k_linproj<_Float16><<<rowBlocksM, 256, 0, stream>>>(att, Wf[4], kocc, MS);
  k_linproj<_Float16><<<rowBlocksM, 256, 0, stream>>>(att, Wf[5], vocc, MS);

  // 6) self init; 7) occurrence scatter; 8) final normalize
  k_self<<<(N + 7) / 8, 256, 0, stream>>>(q2, kslf, vslf, out, N);
  k_occ<<<(MS + 7) / 8, 256, 0, stream>>>(q2, kocc, vocc, subsets, out, MS);
  k_norm<<<(N + 7) / 8, 256, 0, stream>>>(out, N);
}
